// DecoderLayer_32117765439858
// MI455X (gfx1250) — matrix operand order repaired and checked by reading, not compiled, not run on hardware
//
#include <hip/hip_runtime.h>
#include <hip/hip_bf16.h>

// ---- problem constants (match reference) ----
#define BB    2
#define LT    2048
#define LSRC  2048
#define HID   1024
#define HEADS 16
#define HD    64
#define PF    4096
#define EPSLN 1e-5f

typedef __attribute__((ext_vector_type(16))) __bf16 v16bf;
typedef __attribute__((ext_vector_type(8)))  __bf16 v8bf;
typedef __attribute__((ext_vector_type(8)))  float  v8f;
typedef __attribute__((ext_vector_type(4)))  unsigned int v4u;
typedef __attribute__((ext_vector_type(8)))  int v8i_;
typedef __attribute__((ext_vector_type(4)))  int v4i_;

// Assemble the 16x32 bf16 A-operand for one lane in contiguous K order:
// lanes 0-15 hold K=0..15; lanes 16-31 hold K=16..31 (matches B frags).
__device__ __forceinline__ v16bf make_afrag(const __bf16* rowp, int kb8) {
  v8bf lo = *(const v8bf*)(rowp + 2 * kb8);
  v8bf hi = *(const v8bf*)(rowp + 2 * kb8 + 8);
  return __builtin_shufflevector(lo, hi, 0,1,2,3,4,5,6,7,8,9,10,11,12,13,14,15);
}

// Low 32 bits of a generic pointer to __shared__ = LDS byte offset (the LDS
// aperture maps addr[31:0] directly onto the workgroup's LDS allocation).
__device__ __forceinline__ unsigned lds_offset_of(const void* p) {
  return (unsigned)(unsigned long)p;
}

// ---------------------------------------------------------------------------
// Tensor Data Mover: async-copy a 2D tile (tile_x contiguous bf16 elements x
// tile_y rows, row stride = row_stride elements) from global to LDS, letting
// the TDM insert LDS row padding so the tile lands directly in our padded
// fragment layout. D# built per CDNA5 ISA 8.3/8.4. Issue from ONE wave only;
// drain with s_wait_tensorcnt, then barrier for workgroup visibility.
//   pad_interval: pad after (1<<pad_interval) 8-byte units of a row
//   pad_amount:   (pad_amount+1) DWORDs of padding
// ---------------------------------------------------------------------------
__device__ __forceinline__ void tdm_load_tile_2d(
    unsigned lds_byte_off, const __bf16* gsrc,
    unsigned tile_x, unsigned tile_y, unsigned long row_stride,
    unsigned pad_interval, unsigned pad_amount)
{
  unsigned long ga = (unsigned long)(const void*)gsrc;
  v4u g0;
  g0[0] = 1u;                                             // count=1 (user D#)
  g0[1] = lds_byte_off;                                   // lds_addr
  g0[2] = (unsigned)(ga & 0xffffffffu);                   // global_addr[31:0]
  g0[3] = (unsigned)((ga >> 32) & 0x01ffffffu) | (2u << 30); // addr[56:32], type=2
  v8i_ g1;
  g1[0] = (int)((1u << 16) | (1u << 20) |                 // data_size=2B, pad_enable
                (pad_interval << 22) | (pad_amount << 25));
  g1[1] = (int)((tile_x & 0xffffu) << 16);                // tensor_dim0[15:0] @ b48
  g1[2] = (int)((tile_x >> 16) | ((tile_y & 0xffffu) << 16)); // dim0[31:16], dim1[15:0]
  g1[3] = (int)((tile_y >> 16) | ((tile_x & 0xffffu) << 16)); // dim1[31:16], tile_dim0
  g1[4] = (int)(tile_y & 0xffffu);                        // tile_dim1, tile_dim2=0
  g1[5] = (int)(row_stride & 0xffffffffu);                // tensor_dim0_stride[31:0]
  g1[6] = (int)((row_stride >> 32) & 0xffffu);            // stride[47:32], dim1_stride=0
  g1[7] = 0;
  v4i_ z4 = {0, 0, 0, 0};
#if __clang_major__ >= 23
  v8i_ z8 = {0, 0, 0, 0, 0, 0, 0, 0};
  __builtin_amdgcn_tensor_load_to_lds(g0, g1, z4, z4, z8, 0);
#else
  __builtin_amdgcn_tensor_load_to_lds(g0, g1, z4, z4, 0);
#endif
}

// ---------------------------------------------------------------------------
// Generic bf16 GEMM: C[M,N] = A[M,K] * B[K,N] + bias, optional ReLU.
// Block tile 128x128, K-step 32, 8 wave32s arranged 2(M)x4(N), 64x32 per wave.
// A tile staged by the TDM (wave 0), B tile transposed into K-major LDS by all
// waves with batched loads. All dims here are multiples of 128/32.
// ---------------------------------------------------------------------------
__global__ __launch_bounds__(256) void gemm_bf16_128x128(
    const __bf16* __restrict__ A, int lda,
    const __bf16* __restrict__ Bm, int ldb,
    const float*  __restrict__ bias,
    float*  __restrict__ Cf, __bf16* __restrict__ Cbf, int ldc,
    int K, int relu)
{
  __shared__ __align__(64) __bf16 As[128 * 48];   // [m][k], 96B row stride (TDM-padded)
  __shared__ __align__(64) __bf16 BsT[128 * 48];  // [n][k], K-major for B frags

  const int tid  = threadIdx.x;
  const int wave = tid >> 5;
  const int lane = tid & 31;
  const int wm = wave >> 2, wn = wave & 3;
  const long m0 = (long)blockIdx.y * 128;
  const long n0 = (long)blockIdx.x * 128;
  const int l16 = lane & 15;
  const int lh  = lane >> 4;
  const unsigned as_base = lds_offset_of(As);

  v8f acc[4][2] = {};

  for (int k0 = 0; k0 < K; k0 += 32) {
    // A tile 128x32 via TDM: pad after 64B (8 x 8B units -> interval 3) by
    // 8 DWORDs (amount 7) => LDS row stride 48 halves, matching As[].
    if (tid < 32)
      tdm_load_tile_2d(as_base, A + m0 * (long)lda + k0, 32u, 128u,
                       (unsigned long)lda, 3u, 7u);

    // B tile 32x128 -> K-major BsT[n][k]; batch the global loads so they
    // pipeline under a single loadcnt wait, then scatter-transpose.
    uint4 bv[2];
#pragma unroll
    for (int i = 0; i < 2; ++i) {
      int c = tid + i * 256;
      int kk = c >> 4, nn = (c & 15) * 8;
      bv[i] = *(const uint4*)(Bm + (long)(k0 + kk) * ldb + n0 + nn);
    }
#pragma unroll
    for (int i = 0; i < 2; ++i) {
      int c = tid + i * 256;
      int kk = c >> 4, nn = (c & 15) * 8;
      const __bf16* h = (const __bf16*)&bv[i];
#pragma unroll
      for (int j = 0; j < 8; ++j) BsT[(nn + j) * 48 + kk] = h[j];
    }

    if (k0 + 32 < K)  // pull next B K-tile toward L2 (global_prefetch_b8)
      __builtin_prefetch(Bm + (long)(k0 + 32 + (tid >> 4)) * ldb + n0 + (tid & 15) * 8, 0, 1);

    if (tid < 32) __builtin_amdgcn_s_wait_tensorcnt(0);  // drain TDM in wave 0
    __syncthreads();

    v16bf af[4];
#pragma unroll
    for (int mf = 0; mf < 4; ++mf)
      af[mf] = make_afrag(&As[(wm * 64 + mf * 16 + l16) * 48], lh * 8);
#pragma unroll
    for (int nf = 0; nf < 2; ++nf) {
      v16bf bf = *(const v16bf*)&BsT[(wn * 32 + nf * 16 + l16) * 48 + lh * 16];
#pragma unroll
      for (int mf = 0; mf < 4; ++mf)
        acc[mf][nf] = __builtin_amdgcn_wmma_f32_16x16x32_bf16(
            false, af[mf], false, bf, (short)0, acc[mf][nf], false, false);
    }
    __syncthreads();
  }

  const int cmb = lh * 8;
#pragma unroll
  for (int nf = 0; nf < 2; ++nf) {
    long n = n0 + wn * 32 + nf * 16 + l16;
    float bv = bias ? bias[n] : 0.0f;
#pragma unroll
    for (int mf = 0; mf < 4; ++mf) {
#pragma unroll
      for (int r = 0; r < 8; ++r) {
        long m = m0 + wm * 64 + mf * 16 + cmb + r;
        float v = acc[mf][nf][r] + bv;
        if (relu) v = fmaxf(v, 0.0f);
        if (Cf)  Cf[m * (long)ldc + n]  = v;
        if (Cbf) Cbf[m * (long)ldc + n] = (__bf16)v;
      }
    }
  }
}

// ---------------------------------------------------------------------------
// Attention scores: S[bh, q, k] = (Q[b,q,h,:] . K[b,k,h,:]) / 8.
// 128x128 tile per block, d=64 fully staged by two TDM ops (waves 0 and 1),
// 2 WMMA K-steps, grid.z = b*H+h.
// ---------------------------------------------------------------------------
__global__ __launch_bounds__(256) void attn_scores_128x128(
    const __bf16* __restrict__ Qb, const __bf16* __restrict__ Kb,
    float* __restrict__ S)
{
  __shared__ __align__(64) __bf16 Qs[128 * 80];  // [q][d], 160B stride (TDM-padded)
  __shared__ __align__(64) __bf16 Ks[128 * 80];  // [k][d]
  const int tid = threadIdx.x, wave = tid >> 5, lane = tid & 31;
  const int wm = wave >> 2, wn = wave & 3;
  const int l16 = lane & 15, lh = lane >> 4;
  const int bh = blockIdx.z;
  const int b = bh / HEADS, h = bh % HEADS;
  const long m0 = (long)blockIdx.y * 128;
  const long n0 = (long)blockIdx.x * 128;
  const __bf16* Qbase = Qb + ((long)b * LT) * HID + h * HD;
  const __bf16* Kbase = Kb + ((long)b * LSRC) * HID + h * HD;

  // 128 rows x 64 halves each; pad after 128B (16 x 8B -> interval 4) by
  // 8 DWORDs (amount 7) => LDS row stride 80 halves.
  if (tid < 32)
    tdm_load_tile_2d(lds_offset_of(Qs), Qbase + m0 * (long)HID, 64u, 128u,
                     (unsigned long)HID, 4u, 7u);
  else if (tid < 64)
    tdm_load_tile_2d(lds_offset_of(Ks), Kbase + n0 * (long)HID, 64u, 128u,
                     (unsigned long)HID, 4u, 7u);
  if (tid < 64) __builtin_amdgcn_s_wait_tensorcnt(0);
  __syncthreads();

  v8f acc[4][2] = {};
#pragma unroll
  for (int ks = 0; ks < 64; ks += 32) {
    v16bf af[4];
#pragma unroll
    for (int mf = 0; mf < 4; ++mf)
      af[mf] = make_afrag(&Qs[(wm * 64 + mf * 16 + l16) * 80 + ks], lh * 8);
#pragma unroll
    for (int nf = 0; nf < 2; ++nf) {
      v16bf bf = *(const v16bf*)&Ks[(wn * 32 + nf * 16 + l16) * 80 + ks + lh * 16];
#pragma unroll
      for (int mf = 0; mf < 4; ++mf)
        acc[mf][nf] = __builtin_amdgcn_wmma_f32_16x16x32_bf16(
            false, af[mf], false, bf, (short)0, acc[mf][nf], false, false);
    }
  }

  float* Sbase = S + (long)bh * LT * LSRC;
  const int cmb = lh * 8;
#pragma unroll
  for (int nf = 0; nf < 2; ++nf) {
    long n = n0 + wn * 32 + nf * 16 + l16;
#pragma unroll
    for (int mf = 0; mf < 4; ++mf)
#pragma unroll
      for (int r = 0; r < 8; ++r) {
        long m = m0 + wm * 64 + mf * 16 + cmb + r;
        Sbase[m * (long)LSRC + n] = acc[mf][nf][r] * 0.125f;  // 1/sqrt(64)
      }
  }
}

// ---------------------------------------------------------------------------
// Row softmax with mask over 2048 columns, in place. One block per row.
// ---------------------------------------------------------------------------
__global__ __launch_bounds__(256) void softmax_rows_2048(
    float* __restrict__ S, const int* __restrict__ mask)
{
  __shared__ float red[256];
  const int tid = threadIdx.x;
  const long row = blockIdx.x;
  const int b = (int)(row / ((long)HEADS * LT));
  float* p = S + row * (long)LSRC;
  const int* mrow = mask + (long)b * LSRC;

  float vals[8];
  float mx = -3.402823e38f;
#pragma unroll
  for (int i = 0; i < 8; ++i) {
    int j = tid + i * 256;
    float x = p[j];
    vals[i] = (mrow[j] == 0) ? -3.402823e38f : x;
    mx = fmaxf(mx, vals[i]);
  }
  red[tid] = mx; __syncthreads();
  for (int o = 128; o > 0; o >>= 1) { if (tid < o) red[tid] = fmaxf(red[tid], red[tid + o]); __syncthreads(); }
  mx = red[0]; __syncthreads();

  float s = 0.0f;
#pragma unroll
  for (int i = 0; i < 8; ++i) {
    float e = (vals[i] <= -3.0e38f) ? 0.0f : __expf(vals[i] - mx);
    vals[i] = e; s += e;
  }
  red[tid] = s; __syncthreads();
  for (int o = 128; o > 0; o >>= 1) { if (tid < o) red[tid] += red[tid + o]; __syncthreads(); }
  const float inv = (red[0] > 0.0f) ? (1.0f / red[0]) : 0.0f;
#pragma unroll
  for (int i = 0; i < 8; ++i) p[tid + i * 256] = vals[i] * inv;
}

// ---------------------------------------------------------------------------
// O[b,q,h,:] = P[bh] (f32, converted to bf16 in LDS) x V[b,:,h,:] (bf16).
// M tile 128, N=64 full, K-step 32; 8 waves each own a 16x64 stripe.
// P staging cannot use TDM (needs f32->bf16 convert); batched manual loads.
// ---------------------------------------------------------------------------
__global__ __launch_bounds__(256) void attn_av_128x64(
    const float* __restrict__ P, const __bf16* __restrict__ Vb,
    __bf16* __restrict__ Ob)
{
  __shared__ __align__(64) __bf16 Ps[128 * 48];  // [q][k]
  __shared__ __align__(64) __bf16 Vs[64 * 48];   // [d][k]
  const int tid = threadIdx.x, wave = tid >> 5, lane = tid & 31;
  const int l16 = lane & 15, lh = lane >> 4;
  const int bh = blockIdx.y;
  const int b = bh / HEADS, h = bh % HEADS;
  const long m0 = (long)blockIdx.x * 128;
  const float*  Pbase = P + (long)bh * LT * LSRC + m0 * (long)LSRC;
  const __bf16* Vbase = Vb + ((long)b * LSRC) * HID + h * HD;

  v8f acc[4] = {};
  for (int k0 = 0; k0 < LSRC; k0 += 32) {
    float4 pv[4];
#pragma unroll
    for (int i = 0; i < 4; ++i) {
      int c = tid + i * 256;
      int r = c >> 3, cc = (c & 7) * 4;
      pv[i] = *(const float4*)(Pbase + (long)r * LSRC + k0 + cc);
    }
    uint4 vv;
    {
      int kk = tid >> 3, dd = (tid & 7) * 8;
      vv = *(const uint4*)(Vbase + (long)(k0 + kk) * HID + dd);
    }
#pragma unroll
    for (int i = 0; i < 4; ++i) {
      int c = tid + i * 256;
      int r = c >> 3, cc = (c & 7) * 4;
      Ps[r * 48 + cc + 0] = (__bf16)pv[i].x;
      Ps[r * 48 + cc + 1] = (__bf16)pv[i].y;
      Ps[r * 48 + cc + 2] = (__bf16)pv[i].z;
      Ps[r * 48 + cc + 3] = (__bf16)pv[i].w;
    }
    {
      int kk = tid >> 3, dd = (tid & 7) * 8;
      const __bf16* hv = (const __bf16*)&vv;
#pragma unroll
      for (int j = 0; j < 8; ++j) Vs[(dd + j) * 48 + kk] = hv[j];
    }
    __syncthreads();

    v16bf af = make_afrag(&Ps[(wave * 16 + l16) * 48], lh * 8);
#pragma unroll
    for (int nf = 0; nf < 4; ++nf) {
      v16bf bf = *(const v16bf*)&Vs[(nf * 16 + l16) * 48 + lh * 16];
      acc[nf] = __builtin_amdgcn_wmma_f32_16x16x32_bf16(
          false, af, false, bf, (short)0, acc[nf], false, false);
    }
    __syncthreads();
  }

  __bf16* Obase = Ob + ((long)b * LT) * HID + h * HD;
#pragma unroll
  for (int nf = 0; nf < 4; ++nf)
#pragma unroll
    for (int r = 0; r < 8; ++r) {
      long m = m0 + wave * 16 + lh * 8 + r;
      Obase[m * (long)HID + nf * 16 + l16] = (__bf16)acc[nf][r];
    }
}

// ---------------------------------------------------------------------------
// y = LayerNorm(x + t) * g + b ; emits f32 (residual/output) and bf16 copies.
// ---------------------------------------------------------------------------
__global__ __launch_bounds__(256) void add_layernorm_1024(
    const float* __restrict__ X, const float* __restrict__ T,
    const float* __restrict__ g, const float* __restrict__ bt,
    float* __restrict__ Yf, __bf16* __restrict__ Ybf)
{
  __shared__ float red[256];
  const long base = (long)blockIdx.x * HID;
  const int tid = threadIdx.x;
  float v4[4];
  float s = 0.0f;
#pragma unroll
  for (int i = 0; i < 4; ++i) {
    int j = tid * 4 + i;
    v4[i] = X[base + j] + T[base + j];
    s += v4[i];
  }
  red[tid] = s; __syncthreads();
  for (int o = 128; o > 0; o >>= 1) { if (tid < o) red[tid] += red[tid + o]; __syncthreads(); }
  const float mean = red[0] * (1.0f / HID);
  __syncthreads();
  float vs = 0.0f;
#pragma unroll
  for (int i = 0; i < 4; ++i) { float d = v4[i] - mean; vs += d * d; }
  red[tid] = vs; __syncthreads();
  for (int o = 128; o > 0; o >>= 1) { if (tid < o) red[tid] += red[tid + o]; __syncthreads(); }
  const float rstd = rsqrtf(red[0] * (1.0f / HID) + EPSLN);
#pragma unroll
  for (int i = 0; i < 4; ++i) {
    int j = tid * 4 + i;
    float y = (v4[i] - mean) * rstd * g[j] + bt[j];
    Yf[base + j]  = y;
    Ybf[base + j] = (__bf16)y;
  }
}

__global__ void cvt_f32_bf16(const float* __restrict__ src, __bf16* __restrict__ dst, long n) {
  long i = (long)blockIdx.x * blockDim.x + threadIdx.x;
  const long stride = (long)gridDim.x * blockDim.x;
  for (; i < n; i += stride) dst[i] = (__bf16)src[i];
}

// ---------------------------------------------------------------------------
extern "C" void kernel_launch(void* const* d_in, const int* in_sizes, int n_in,
                              void* d_out, int out_size, void* d_ws, size_t ws_size,
                              hipStream_t stream) {
  (void)in_sizes; (void)n_in; (void)out_size; (void)ws_size;
  const float* tgt   = (const float*)d_in[0];
  const float* enc   = (const float*)d_in[1];
  const int*   tmask = (const int*)d_in[2];
  const int*   smask = (const int*)d_in[3];
  const float* sa_w[4] = {(const float*)d_in[4], (const float*)d_in[6], (const float*)d_in[8], (const float*)d_in[10]};
  const float* sa_b[4] = {(const float*)d_in[5], (const float*)d_in[7], (const float*)d_in[9], (const float*)d_in[11]};
  const float* ea_w[4] = {(const float*)d_in[12], (const float*)d_in[14], (const float*)d_in[16], (const float*)d_in[18]};
  const float* ea_b[4] = {(const float*)d_in[13], (const float*)d_in[15], (const float*)d_in[17], (const float*)d_in[19]};
  const float* ffn_w1 = (const float*)d_in[20]; const float* ffn_b1 = (const float*)d_in[21];
  const float* ffn_w2 = (const float*)d_in[22]; const float* ffn_b2 = (const float*)d_in[23];
  const float* ln1_g = (const float*)d_in[24]; const float* ln1_b = (const float*)d_in[25];
  const float* ln2_g = (const float*)d_in[26]; const float* ln2_b = (const float*)d_in[27];
  const float* ln3_g = (const float*)d_in[28]; const float* ln3_b = (const float*)d_in[29];

  // ---- workspace carve (~672 MB) ----
  char* wsp = (char*)d_ws; size_t off = 0;
  auto take = [&](size_t bytes) -> void* {
    void* r = wsp + off; off += (bytes + 255) & ~(size_t)255; return r;
  };
  const long NTOK = (long)BB * LT;            // 4096 rows
  const long NACT = NTOK * HID;               // 4,194,304
  __bf16* wbf_sa[4]; for (int i = 0; i < 4; ++i) wbf_sa[i] = (__bf16*)take((size_t)HID * HID * 2);
  __bf16* wbf_ea[4]; for (int i = 0; i < 4; ++i) wbf_ea[i] = (__bf16*)take((size_t)HID * HID * 2);
  __bf16* w1_bf  = (__bf16*)take((size_t)HID * PF * 2);
  __bf16* w2_bf  = (__bf16*)take((size_t)PF * HID * 2);
  __bf16* cur_bf = (__bf16*)take((size_t)NACT * 2);
  __bf16* enc_bf = (__bf16*)take((size_t)NACT * 2);
  __bf16* q_bf   = (__bf16*)take((size_t)NACT * 2);
  __bf16* k_bf   = (__bf16*)take((size_t)NACT * 2);
  __bf16* v_bf   = (__bf16*)take((size_t)NACT * 2);
  __bf16* o_bf   = (__bf16*)take((size_t)NACT * 2);
  __bf16* h1_bf  = (__bf16*)take((size_t)NTOK * PF * 2);
  float*  t_f32  = (float*)take((size_t)NACT * 4);
  float*  resid1 = (float*)take((size_t)NACT * 4);
  float*  resid2 = (float*)take((size_t)NACT * 4);
  float*  scores = (float*)take((size_t)BB * HEADS * LT * LSRC * 4);   // 512 MB

  float* out_tgt  = (float*)d_out;
  float* out_attn = (float*)d_out + (long)BB * LT * HID;

  auto cvt = [&](const float* s, __bf16* d, long n) {
    cvt_f32_bf16<<<4096, 256, 0, stream>>>(s, d, n);
  };
  auto gemm = [&](const __bf16* A, int lda, const __bf16* Bm, int ldb,
                  const float* bias, float* Cf, __bf16* Cbf, int ldc,
                  int M, int N, int K, int relu) {
    dim3 g(N / 128, M / 128);
    gemm_bf16_128x128<<<g, 256, 0, stream>>>(A, lda, Bm, ldb, bias, Cf, Cbf, ldc, K, relu);
  };
  const dim3 att_g(LSRC / 128, LT / 128, BB * HEADS);
  const dim3 av_g(LT / 128, BB * HEADS);
  const int  sm_rows = BB * HEADS * LT;

  // ---- one-time precision conversions ----
  for (int i = 0; i < 4; ++i) cvt(sa_w[i], wbf_sa[i], (long)HID * HID);
  for (int i = 0; i < 4; ++i) cvt(ea_w[i], wbf_ea[i], (long)HID * HID);
  cvt(ffn_w1, w1_bf, (long)HID * PF);
  cvt(ffn_w2, w2_bf, (long)PF * HID);
  cvt(tgt, cur_bf, NACT);
  cvt(enc, enc_bf, NACT);

  // ---- self-attention ----
  gemm(cur_bf, HID, wbf_sa[0], HID, sa_b[0], nullptr, q_bf, HID, (int)NTOK, HID, HID, 0);
  gemm(cur_bf, HID, wbf_sa[1], HID, sa_b[1], nullptr, k_bf, HID, (int)NTOK, HID, HID, 0);
  gemm(cur_bf, HID, wbf_sa[2], HID, sa_b[2], nullptr, v_bf, HID, (int)NTOK, HID, HID, 0);
  attn_scores_128x128<<<att_g, 256, 0, stream>>>(q_bf, k_bf, scores);
  softmax_rows_2048<<<sm_rows, 256, 0, stream>>>(scores, tmask);
  attn_av_128x64<<<av_g, 256, 0, stream>>>(scores, v_bf, o_bf);
  gemm(o_bf, HID, wbf_sa[3], HID, sa_b[3], t_f32, nullptr, HID, (int)NTOK, HID, HID, 0);
  add_layernorm_1024<<<(int)NTOK, 256, 0, stream>>>(tgt, t_f32, ln1_g, ln1_b, resid1, cur_bf);

  // ---- cross-attention (probs land directly in d_out) ----
  gemm(cur_bf, HID, wbf_ea[0], HID, ea_b[0], nullptr, q_bf, HID, (int)NTOK, HID, HID, 0);
  gemm(enc_bf, HID, wbf_ea[1], HID, ea_b[1], nullptr, k_bf, HID, (int)NTOK, HID, HID, 0);
  gemm(enc_bf, HID, wbf_ea[2], HID, ea_b[2], nullptr, v_bf, HID, (int)NTOK, HID, HID, 0);
  attn_scores_128x128<<<att_g, 256, 0, stream>>>(q_bf, k_bf, out_attn);
  softmax_rows_2048<<<sm_rows, 256, 0, stream>>>(out_attn, smask);
  attn_av_128x64<<<av_g, 256, 0, stream>>>(out_attn, v_bf, o_bf);
  gemm(o_bf, HID, wbf_ea[3], HID, ea_b[3], t_f32, nullptr, HID, (int)NTOK, HID, HID, 0);
  add_layernorm_1024<<<(int)NTOK, 256, 0, stream>>>(resid1, t_f32, ln2_g, ln2_b, resid2, cur_bf);

  // ---- position-wise FFN ----
  gemm(cur_bf, HID, w1_bf, PF, ffn_b1, nullptr, h1_bf, PF, (int)NTOK, PF, HID, 1);
  gemm(h1_bf, PF, w2_bf, HID, ffn_b2, t_f32, nullptr, HID, (int)NTOK, HID, PF, 0);
  add_layernorm_1024<<<(int)NTOK, 256, 0, stream>>>(resid2, t_f32, ln3_g, ln3_b, out_tgt, cur_bf);
}
